// Attention_67302137528533
// MI455X (gfx1250) — matrix operand order repaired and checked by reading, not compiled, not run on hardware
//
#include <hip/hip_runtime.h>
#include <stdint.h>

// ---------- types ----------
typedef __bf16 bf16;
typedef __attribute__((ext_vector_type(16))) __bf16 v16bf;
typedef __attribute__((ext_vector_type(8)))  __bf16 v8bf;
typedef __attribute__((ext_vector_type(8)))  float  v8f;

#define D_MODEL   1024
#define NUM_HEADS 16
#define HEAD_DIM  64
#define SEQ       2048
#define BATCH     2
#define MROWS     (BATCH * SEQ)   // 4096

// ---------- helpers ----------
__device__ __forceinline__ bf16 f2bf(float f) {
  union { float f; uint32_t u; } a; a.f = f;
  uint32_t r = a.u + 0x7FFFu + ((a.u >> 16) & 1u);   // round-to-nearest-even
  union { uint16_t s; bf16 b; } c; c.s = (uint16_t)(r >> 16);
  return c.b;
}

__device__ __forceinline__ v16bf load_frag(const bf16* p0, const bf16* p1) {
  v8bf lo = *(const v8bf*)p0;
  v8bf hi = *(const v8bf*)p1;
  v16bf r;
#pragma unroll
  for (int i = 0; i < 8; ++i) { r[i] = lo[i]; r[i + 8] = hi[i]; }
  return r;
}

__device__ __forceinline__ v8f zero8() {
  v8f z;
#pragma unroll
  for (int i = 0; i < 8; ++i) z[i] = 0.0f;
  return z;
}

__device__ __forceinline__ v8f wmma_bf16(v16bf a, v16bf b, v8f c) {
  // D = A(16x32) * B(32x16) + C, f32 accumulate
  return __builtin_amdgcn_wmma_f32_16x16x32_bf16(false, a, false, b,
                                                 (short)0, c, false, false);
}

// ---- CDNA5 async copy: 32 bytes global -> LDS per lane, no VGPR round-trip.
// (offset applies to BOTH addresses, so one address pair covers both halves).
__device__ __forceinline__ void async_b128_x2(void* lds, const void* gsrc) {
  asm volatile(
      "global_load_async_to_lds_b128 %0, %1, off\n\t"
      "global_load_async_to_lds_b128 %0, %1, off offset:16"
      :
      : "v"((uint32_t)(uintptr_t)lds),
        "v"((unsigned long long)(uintptr_t)gsrc)
      : "memory");
}

__device__ __forceinline__ void wait_async0() {
#if __has_builtin(__builtin_amdgcn_s_wait_asynccnt)
  __builtin_amdgcn_s_wait_asynccnt(0);
#else
  asm volatile("s_wait_asynccnt 0x0" ::: "memory");
#endif
}

// ---------- stage 0: fp32 -> bf16 ----------
__global__ void k_cvt_bf16(const float* __restrict__ src, bf16* __restrict__ dst, int n) {
  int i = blockIdx.x * blockDim.x + threadIdx.x;
  if (i < n) dst[i] = f2bf(src[i]);
}

// ---------- stage 0b: transpose weight fp32[n][n] -> bf16 WT[o][d] ----------
__global__ void k_transpose_bf16(const float* __restrict__ W, bf16* __restrict__ WT, int n) {
  __shared__ float tile[32][33];
  int bx = blockIdx.x * 32, by = blockIdx.y * 32;
  int tx = threadIdx.x, ty = threadIdx.y;          // block (32, 8)
#pragma unroll
  for (int i = 0; i < 32; i += 8)
    tile[ty + i][tx] = W[(size_t)(by + ty + i) * n + bx + tx];
  __syncthreads();
#pragma unroll
  for (int i = 0; i < 32; i += 8)
    WT[(size_t)(bx + ty + i) * n + by + tx] = f2bf(tile[tx][ty + i]);
}

// ---------- stage 1/4: bf16 WMMA GEMM  C = A[MxK] * BT[NxK]^T + bias ----------
// Async double-buffered LDS staging: tile copy via global_load_async_to_lds_b128
// overlaps the next K-slice's memory traffic with the current slice's WMMAs.
// mode 0: fp32 out [MxN]         (final projection)
// mode 1: bf16 out, head-split   [B,H,S,64]   (Q and K)
// mode 2: bf16 out, head-split T [B,H,64,S]   (V transposed)
#define BM 128
#define BN 64
#define BK 32

__global__ __launch_bounds__(256)
void k_gemm_bf16(const bf16* __restrict__ A, const bf16* __restrict__ BT,
                 const float* __restrict__ bias, void* __restrict__ out,
                 int M, int N, int K, int mode) {
  __shared__ bf16 sA[2][BM * BK];
  __shared__ bf16 sB[2][BN * BK];

  const int tid  = threadIdx.x;
  const int lane = tid & 31;
  const int wave = tid >> 5;        // 8 waves: 4 (M) x 2 (N)
  const int wm   = wave & 3;
  const int wn   = wave >> 2;
  const int r    = lane & 15;
  const int h    = lane >> 4;
  const int bm   = blockIdx.x * BM;
  const int bn   = blockIdx.y * BN;

  // staging slice owned by this thread: 2 threads per tile row, 32B each
  const int row  = tid >> 1;               // 0..127
  const int half = (tid & 1) * 16;         // element offset within row
  const bf16* gA = A  + (size_t)(bm + row) * K + half;
  const bf16* gB = BT + (size_t)(bn + (row & (BN - 1))) * K + half;

  v8f acc[2][2];
#pragma unroll
  for (int mi = 0; mi < 2; ++mi)
#pragma unroll
    for (int ni = 0; ni < 2; ++ni) acc[mi][ni] = zero8();

  // prologue: stage k0 = 0 into buffer 0
  async_b128_x2(&sA[0][row * BK + half], gA);
  if (tid < 2 * BN) async_b128_x2(&sB[0][row * BK + half], gB);

  int cur = 0;
  for (int k0 = 0; k0 < K; k0 += BK) {
    wait_async0();        // this wave's async LDS writes are done
    __syncthreads();      // ... and so is everyone else's

    if (k0 + BK < K) {    // issue next slice into the other buffer (no wait)
      async_b128_x2(&sA[cur ^ 1][row * BK + half], gA + k0 + BK);
      if (tid < 2 * BN) async_b128_x2(&sB[cur ^ 1][row * BK + half], gB + k0 + BK);
    }

    v16bf af[2], bfr[2];
#pragma unroll
    for (int mi = 0; mi < 2; ++mi) {
      const bf16* p = sA[cur] + (size_t)(wm * 32 + mi * 16 + r) * BK;
      af[mi] = load_frag(p + h * 16, p + h * 16 + 8);      // k = 16h + s, same order as B
    }
#pragma unroll
    for (int ni = 0; ni < 2; ++ni) {
      const bf16* p = sB[cur] + (size_t)(wn * 32 + ni * 16 + r) * BK + h * 16;
      bfr[ni] = load_frag(p, p + 8);                       // k = 16h + s (contig in B^T)
    }
#pragma unroll
    for (int mi = 0; mi < 2; ++mi)
#pragma unroll
      for (int ni = 0; ni < 2; ++ni)
        acc[mi][ni] = wmma_bf16(af[mi], bfr[ni], acc[mi][ni]);

    cur ^= 1;
  }

  // epilogue: D layout -> lane(0-15): row i, lane(16-31): row 8+i ; col = lane&15
#pragma unroll
  for (int mi = 0; mi < 2; ++mi)
#pragma unroll
    for (int ni = 0; ni < 2; ++ni)
#pragma unroll
      for (int i = 0; i < 8; ++i) {
        int gm = bm + wm * 32 + mi * 16 + h * 8 + i;
        int gn = bn + wn * 32 + ni * 16 + r;
        float v = acc[mi][ni][i] + bias[gn];
        if (mode == 0) {
          ((float*)out)[(size_t)gm * N + gn] = v;
        } else {
          int b  = gm >> 11, s  = gm & (SEQ - 1);          // S = 2048
          int hh = gn >> 6,  dh = gn & (HEAD_DIM - 1);
          size_t dst = (mode == 1)
              ? ((size_t)(b * NUM_HEADS + hh) * SEQ + s) * HEAD_DIM + dh
              : ((size_t)(b * NUM_HEADS + hh) * HEAD_DIM + dh) * SEQ + s;
          ((bf16*)out)[dst] = f2bf(v);
        }
      }
}

// ---------- stage 2: flash attention (bf16 WMMA, online softmax) ----------
#define QB 64
#define KBLK 64

__global__ __launch_bounds__(128)
void k_attention(const bf16* __restrict__ Q,   // [B*H, S, 64]
                 const bf16* __restrict__ Kk,  // [B*H, S, 64]
                 const bf16* __restrict__ Vt,  // [B*H, 64, S]
                 bf16* __restrict__ O) {       // [B*S, 1024]
  const int bh   = blockIdx.y;                 // 0..31
  const int b    = bh >> 4, hh = bh & 15;
  const int q0   = blockIdx.x * QB;
  const int tid  = threadIdx.x;
  const int lane = tid & 31, wave = tid >> 5;  // 4 waves x 16 query rows
  const int r    = lane & 15, h = lane >> 4;
  const int qrow = q0 + wave * 16;

  __shared__ bf16 sP[4][16 * 64];              // per-wave P tile (bf16)

  const bf16* qbase = Q  + ((size_t)bh * SEQ + qrow) * HEAD_DIM;
  const bf16* kbase = Kk + (size_t)bh * SEQ * HEAD_DIM;
  const bf16* vbase = Vt + (size_t)bh * HEAD_DIM * SEQ;

  // Q fragments held in registers (2 k-chunks of 32 over hd=64)
  v16bf qf[2];
#pragma unroll
  for (int kc = 0; kc < 2; ++kc) {
    const bf16* p = qbase + (size_t)r * HEAD_DIM + kc * 32;
    qf[kc] = load_frag(p + h * 16, p + h * 16 + 8);
  }

  v8f oacc[4];
#pragma unroll
  for (int t = 0; t < 4; ++t) oacc[t] = zero8();
  float mrow[8], lrow[8];
#pragma unroll
  for (int i = 0; i < 8; ++i) { mrow[i] = -3.0e38f; lrow[i] = 0.0f; }

  const float scale = 0.125f;  // 1/sqrt(64)

  for (int kb = 0; kb < SEQ; kb += KBLK) {
    if (kb + KBLK < SEQ)   // hint next K block into cache (global_prefetch_b8)
      __builtin_prefetch(kbase + (size_t)(kb + KBLK) * HEAD_DIM + tid * 32, 0, 0);

    // ---- scores S = Q * K^T : B-fragment = rows of K (contiguous) ----
    v8f sacc[4];
#pragma unroll
    for (int t = 0; t < 4; ++t) sacc[t] = zero8();
#pragma unroll
    for (int nt = 0; nt < 4; ++nt)
#pragma unroll
      for (int kc = 0; kc < 2; ++kc) {
        const bf16* p = kbase + (size_t)(kb + nt * 16 + r) * HEAD_DIM + kc * 32 + h * 16;
        sacc[nt] = wmma_bf16(qf[kc], load_frag(p, p + 8), sacc[nt]);
      }

    // ---- online softmax: rows split lanes 0-15 (i) / 16-31 (8+i) ----
    float corr[8];
#pragma unroll
    for (int i = 0; i < 8; ++i) {
      float mx = fmaxf(fmaxf(sacc[0][i], sacc[1][i]), fmaxf(sacc[2][i], sacc[3][i])) * scale;
#pragma unroll
      for (int mk = 1; mk < 16; mk <<= 1) mx = fmaxf(mx, __shfl_xor(mx, mk, 32));
      float mnew = fmaxf(mrow[i], mx);
      corr[i] = __expf(mrow[i] - mnew);
      mrow[i] = mnew;
    }
#pragma unroll
    for (int i = 0; i < 8; ++i) {
      const int m = i + 8 * h;
      float rs = 0.0f;
#pragma unroll
      for (int nt = 0; nt < 4; ++nt) {
        float pv = __expf(sacc[nt][i] * scale - mrow[i]);
        rs += pv;
        sP[wave][m * 64 + nt * 16 + r] = f2bf(pv);
      }
#pragma unroll
      for (int mk = 1; mk < 16; mk <<= 1) rs += __shfl_xor(rs, mk, 32);
      lrow[i] = lrow[i] * corr[i] + rs;
#pragma unroll
      for (int nt = 0; nt < 4; ++nt) oacc[nt][i] *= corr[i];
    }

    // ---- O += P * V : A = P from LDS, B-fragment = rows of V^T (contiguous) ----
    v16bf pf[2];
#pragma unroll
    for (int kc = 0; kc < 2; ++kc) {
      const bf16* p = sP[wave] + (size_t)r * 64 + kc * 32;
      pf[kc] = load_frag(p + h * 16, p + h * 16 + 8);
    }
#pragma unroll
    for (int nt = 0; nt < 4; ++nt)
#pragma unroll
      for (int kc = 0; kc < 2; ++kc) {
        const bf16* p = vbase + (size_t)(nt * 16 + r) * SEQ + kb + kc * 32 + h * 16;
        oacc[nt] = wmma_bf16(pf[kc], load_frag(p, p + 8), oacc[nt]);
      }
  }

  // ---- normalize + store O[b][s][h*64+dh] (concat-heads, row-major) ----
#pragma unroll
  for (int nt = 0; nt < 4; ++nt)
#pragma unroll
    for (int i = 0; i < 8; ++i) {
      const int m = i + 8 * h;
      float v = oacc[nt][i] / lrow[i];
      O[((size_t)(b * SEQ + qrow + m)) * D_MODEL + hh * HEAD_DIM + nt * 16 + r] = f2bf(v);
    }
}

// ---------- launch ----------
extern "C" void kernel_launch(void* const* d_in, const int* in_sizes, int n_in,
                              void* d_out, int out_size, void* d_ws, size_t ws_size,
                              hipStream_t stream) {
  (void)in_sizes; (void)n_in; (void)out_size; (void)ws_size;
  const float* x  = (const float*)d_in[0];
  const float* Wq = (const float*)d_in[1];
  const float* bq = (const float*)d_in[2];
  const float* Wk = (const float*)d_in[3];
  const float* bk = (const float*)d_in[4];
  const float* Wv = (const float*)d_in[5];
  const float* bv = (const float*)d_in[6];
  const float* Wo = (const float*)d_in[7];
  const float* bo = (const float*)d_in[8];

  char* ws = (char*)d_ws;
  const size_t MB = 1u << 20;
  bf16* xb  = (bf16*)(ws + 0 * MB);    // 8 MB: x in bf16          [4096 x 1024]
  bf16* WqT = (bf16*)(ws + 8 * MB);    // 2 MB each: transposed bf16 weights
  bf16* WkT = (bf16*)(ws + 10 * MB);
  bf16* WvT = (bf16*)(ws + 12 * MB);
  bf16* WoT = (bf16*)(ws + 14 * MB);
  bf16* Qb  = (bf16*)(ws + 16 * MB);   // 8 MB [B,H,S,64]
  bf16* Kb  = (bf16*)(ws + 24 * MB);   // 8 MB [B,H,S,64]
  bf16* Vtb = (bf16*)(ws + 32 * MB);   // 8 MB [B,H,64,S]
  bf16* Ob  = (bf16*)(ws + 40 * MB);   // 8 MB [B*S, 1024]

  const int nElem = MROWS * D_MODEL;   // 4 M

  k_cvt_bf16<<<nElem / 256, 256, 0, stream>>>(x, xb, nElem);

  dim3 tb(32, 8), tg(D_MODEL / 32, D_MODEL / 32);
  k_transpose_bf16<<<tg, tb, 0, stream>>>(Wq, WqT, D_MODEL);
  k_transpose_bf16<<<tg, tb, 0, stream>>>(Wk, WkT, D_MODEL);
  k_transpose_bf16<<<tg, tb, 0, stream>>>(Wv, WvT, D_MODEL);
  k_transpose_bf16<<<tg, tb, 0, stream>>>(Wo, WoT, D_MODEL);

  dim3 gg(MROWS / BM, D_MODEL / BN);   // (32, 16)
  k_gemm_bf16<<<gg, 256, 0, stream>>>(xb, WqT, bq, Qb,  MROWS, D_MODEL, D_MODEL, 1);
  k_gemm_bf16<<<gg, 256, 0, stream>>>(xb, WkT, bk, Kb,  MROWS, D_MODEL, D_MODEL, 1);
  k_gemm_bf16<<<gg, 256, 0, stream>>>(xb, WvT, bv, Vtb, MROWS, D_MODEL, D_MODEL, 2);

  dim3 ag(SEQ / QB, BATCH * NUM_HEADS);   // (32, 32)
  k_attention<<<ag, 128, 0, stream>>>(Qb, Kb, Vtb, Ob);

  k_gemm_bf16<<<gg, 256, 0, stream>>>(Ob, WoT, bo, d_out, MROWS, D_MODEL, D_MODEL, 0);
}
